// DHSMBlock_37014028157379
// MI455X (gfx1250) — compile-verified, matrix operand order checked by reading, not run on hardware
//
#include <hip/hip_runtime.h>
#include <hip/hip_bf16.h>

typedef __attribute__((ext_vector_type(16))) _Float16 v16h;
typedef __attribute__((ext_vector_type(8)))  _Float16 v8h;
typedef __attribute__((ext_vector_type(8)))  float    v8f;

#define BSZ 8
#define TT  1024
#define HH  1024
#define SS  128
#define NHH 16
#define RHH 4
#define BT  (BSZ * TT)   // 8192 rows

// LDS row stride (halves) with +8 padding to break bank conflicts
#define APAD 72
#define BPAD 72

// ---------------------------------------------------------------------------
// Activation epilogue: 0=none 1=relu 2=gelu(exact) 3=sigmoid
// ---------------------------------------------------------------------------
__device__ __forceinline__ float act_apply(float v, int act) {
  switch (act) {
    case 1: return fmaxf(v, 0.0f);
    case 2: return 0.5f * v * (1.0f + erff(v * 0.7071067811865476f));
    case 3: return 1.0f / (1.0f + expf(-v));
    default: return v;
  }
}

__device__ __forceinline__ v16h make_op(const _Float16* lo, const _Float16* hi) {
  v8h a = *(const v8h*)lo;
  v8h b = *(const v8h*)hi;
  return __builtin_shufflevector(a, b, 0,1,2,3,4,5,6,7,8,9,10,11,12,13,14,15);
}

// ---------------------------------------------------------------------------
// Batched WMMA GEMM:  C[z] = act( alpha * A[z] @ op(B[z]) + bias ) (+= optional)
//   A: M x K row-major (stride lda), per-batch offset (z/inner)*sAo + (z%inner)*sAi
//   TRANSB==0: B is N x K row-major  (computes A @ B^T  — matches x @ w.T)
//   TRANSB==1: B is K x N row-major  (computes A @ B    — used for att @ V)
// Block tile 128x64, BK=64 staged; 8 waves in 4x2 grid, each wave owns a
// 32x32 C sub-tile => 8 v_wmma_f32_16x16x32_f16 per staged k-iteration.
// Requires M % 128 == 0, N % 64 == 0, K % 64 == 0 (true for every call here).
// ---------------------------------------------------------------------------
template <int TRANSB>
__global__ __launch_bounds__(256)
void gemm_wmma_kernel(const float* __restrict__ A, int lda, long long sAo, long long sAi,
                      const float* __restrict__ B, int ldb, long long sBo, long long sBi,
                      float* __restrict__ C, int ldc, long long sCo, long long sCi,
                      const float* __restrict__ bias,
                      int M, int N, int K, int innerCount,
                      float alpha, int act, int accum)
{
  __shared__ __align__(16) _Float16 As[128][APAD];
  __shared__ __align__(16) _Float16 Bs[64][BPAD];

  const int tid  = threadIdx.x;
  const int lane = tid & 31;
  const int wave = tid >> 5;
  const int wm   = wave >> 1;          // 0..3 : 32-row strip
  const int wn   = wave & 1;           // 0..1 : 32-col strip

  const int z = blockIdx.z;
  const long long offA = (long long)(z / innerCount) * sAo + (long long)(z % innerCount) * sAi;
  const long long offB = (long long)(z / innerCount) * sBo + (long long)(z % innerCount) * sBi;
  const long long offC = (long long)(z / innerCount) * sCo + (long long)(z % innerCount) * sCi;
  const float* Ab = A + offA;
  const float* Bb = B + offB;
  float*       Cb = C + offC;

  const int m0 = blockIdx.y * 128;
  const int n0 = blockIdx.x * 64;

  // cooperative staging indices (8 contiguous floats per access)
  const int srow  = tid >> 3;          // 0..31
  const int scol8 = (tid & 7) << 3;    // 0,8,...,56

  // WMMA 16-bit operand lane mapping (ISA 7.12.2)
  const int mr = lane & 15;
  const int kb = (lane >> 4) << 3;     // 0 or 8

  v8f acc00 = {}, acc01 = {}, acc10 = {}, acc11 = {};

  for (int k0 = 0; k0 < K; k0 += 64) {
    __syncthreads();
    // ---- stage A: 128 x 64 floats -> f16 LDS (4 passes of 32 rows) ----
#pragma unroll
    for (int p = 0; p < 4; ++p) {
      const int row = srow + p * 32;
      const float* ap = Ab + (long long)(m0 + row) * lda + (k0 + scol8);
      float4 f0 = ((const float4*)ap)[0];
      float4 f1 = ((const float4*)ap)[1];
      if (p == 0 && k0 + 64 < K) __builtin_prefetch(ap + 64, 0, 1);
      v8h ah;
      ah[0] = (_Float16)f0.x; ah[1] = (_Float16)f0.y;
      ah[2] = (_Float16)f0.z; ah[3] = (_Float16)f0.w;
      ah[4] = (_Float16)f1.x; ah[5] = (_Float16)f1.y;
      ah[6] = (_Float16)f1.z; ah[7] = (_Float16)f1.w;
      *(v8h*)&As[row][scol8] = ah;
    }
    // ---- stage B: 64 x 64 (N x K layout in LDS) ----
    if (TRANSB == 0) {
#pragma unroll
      for (int p = 0; p < 2; ++p) {
        const int row = srow + p * 32;
        const float* bp = Bb + (long long)(n0 + row) * ldb + (k0 + scol8);
        float4 f0 = ((const float4*)bp)[0];
        float4 f1 = ((const float4*)bp)[1];
        if (p == 0 && k0 + 64 < K) __builtin_prefetch(bp + 64, 0, 1);
        v8h bh;
        bh[0] = (_Float16)f0.x; bh[1] = (_Float16)f0.y;
        bh[2] = (_Float16)f0.z; bh[3] = (_Float16)f0.w;
        bh[4] = (_Float16)f1.x; bh[5] = (_Float16)f1.y;
        bh[6] = (_Float16)f1.z; bh[7] = (_Float16)f1.w;
        *(v8h*)&Bs[row][scol8] = bh;
      }
    } else {
      // B is K x N: gather a column per thread group (transpose into LDS)
      const int nn = tid & 63;          // 0..63
      const int kg = (tid >> 6) << 4;   // 0,16,32,48
#pragma unroll
      for (int j = 0; j < 16; ++j)
        Bs[nn][kg + j] = (_Float16)Bb[(long long)(k0 + kg + j) * ldb + (n0 + nn)];
    }
    __syncthreads();

    // ---- compute: 2 k-steps x (2 m-tiles x 2 n-tiles) = 8 WMMAs ----
#pragma unroll
    for (int ks = 0; ks < 64; ks += 32) {
      const int ra0 = wm * 32 + mr;
      const int ra1 = wm * 32 + 16 + mr;
      const int rb0 = wn * 32 + mr;
      const int rb1 = wn * 32 + 16 + mr;
      v16h a0 = make_op(&As[ra0][ks + kb], &As[ra0][ks + 16 + kb]);
      v16h a1 = make_op(&As[ra1][ks + kb], &As[ra1][ks + 16 + kb]);
      v16h b0 = make_op(&Bs[rb0][ks + kb], &Bs[rb0][ks + 16 + kb]);
      v16h b1 = make_op(&Bs[rb1][ks + kb], &Bs[rb1][ks + 16 + kb]);
      acc00 = __builtin_amdgcn_wmma_f32_16x16x32_f16(false, a0, false, b0, (short)0, acc00, false, false);
      acc01 = __builtin_amdgcn_wmma_f32_16x16x32_f16(false, a0, false, b1, (short)0, acc01, false, false);
      acc10 = __builtin_amdgcn_wmma_f32_16x16x32_f16(false, a1, false, b0, (short)0, acc10, false, false);
      acc11 = __builtin_amdgcn_wmma_f32_16x16x32_f16(false, a1, false, b1, (short)0, acc11, false, false);
    }
  }

  // ---- epilogue ----
  // C/D layout: VGPR r -> lanes 0-15: M=r, N=lane ; lanes 16-31: M=r+8, N=lane-16
  const int nloc  = lane & 15;
  const int mhalf = (lane >> 4) << 3;
  v8f* accs[4] = {&acc00, &acc01, &acc10, &acc11};
#pragma unroll
  for (int ti = 0; ti < 2; ++ti) {
#pragma unroll
    for (int tj = 0; tj < 2; ++tj) {
      v8f& a = *accs[ti * 2 + tj];
      const int n = n0 + wn * 32 + tj * 16 + nloc;
      const float bvv = bias ? bias[n] : 0.0f;
      const int mbase = m0 + wm * 32 + ti * 16 + mhalf;
#pragma unroll
      for (int r = 0; r < 8; ++r) {
        float v = a[r] * alpha + bvv;
        v = act_apply(v, act);
        float* cp = Cb + (long long)(mbase + r) * ldc + n;
        if (accum) v += *cp;
        *cp = v;
      }
    }
  }
}

// ---------------------------------------------------------------------------
// SSM scan: one thread per (batch, state-channel); serial over T.
// ---------------------------------------------------------------------------
__global__ void scan_kernel(const float* __restrict__ gate, const float* __restrict__ bx,
                            const float* __restrict__ A_log, float* __restrict__ states)
{
  int idx = blockIdx.x * blockDim.x + threadIdx.x;
  if (idx >= BSZ * SS) return;
  int b = idx / SS, s = idx % SS;
  float a = expf(A_log[s]);
  const float* gp = gate   + (long long)b * TT * SS + s;
  const float* bp = bx     + (long long)b * TT * SS + s;
  float*       sp = states + (long long)b * TT * SS + s;
  float st = 0.0f;
  for (int t = 0; t < TT; ++t) {
    st = a * st + gp[(long long)t * SS] * bp[(long long)t * SS];
    sp[(long long)t * SS] = st;
  }
}

// ---------------------------------------------------------------------------
// LayerNorm over H=1024 of (a + b); b may be null. One block per row.
// ---------------------------------------------------------------------------
__global__ __launch_bounds__(256)
void ln_kernel(const float* __restrict__ a, const float* __restrict__ b,
               const float* __restrict__ g, const float* __restrict__ be,
               float* __restrict__ out)
{
  const long long row = blockIdx.x;
  const int tid = threadIdx.x;
  const float* ap = a + row * HH;
  const float* bp = b ? b + row * HH : nullptr;
  float x[4];
#pragma unroll
  for (int j = 0; j < 4; ++j) {
    int i = tid + j * 256;
    x[j] = ap[i] + (bp ? bp[i] : 0.0f);
  }
  __shared__ float red[8];
  float s = x[0] + x[1] + x[2] + x[3];
#pragma unroll
  for (int o = 16; o > 0; o >>= 1) s += __shfl_xor(s, o, 32);
  if ((tid & 31) == 0) red[tid >> 5] = s;
  __syncthreads();
  float mean = 0.0f;
#pragma unroll
  for (int w = 0; w < 8; ++w) mean += red[w];
  mean *= (1.0f / HH);
  __syncthreads();
  float vv = 0.0f;
#pragma unroll
  for (int j = 0; j < 4; ++j) { float d = x[j] - mean; vv += d * d; }
#pragma unroll
  for (int o = 16; o > 0; o >>= 1) vv += __shfl_xor(vv, o, 32);
  if ((tid & 31) == 0) red[tid >> 5] = vv;
  __syncthreads();
  float var = 0.0f;
#pragma unroll
  for (int w = 0; w < 8; ++w) var += red[w];
  var *= (1.0f / HH);
  float rs = rsqrtf(var + 1e-5f);
#pragma unroll
  for (int j = 0; j < 4; ++j) {
    int i = tid + j * 256;
    out[row * HH + i] = (x[j] - mean) * rs * g[i] + be[i];
  }
}

// ---------------------------------------------------------------------------
// Row softmax, in place. One block (8 waves) per row of length L.
// ---------------------------------------------------------------------------
__global__ __launch_bounds__(256)
void softmax_kernel(float* __restrict__ data, int L)
{
  const long long row = blockIdx.x;
  float* p = data + row * (long long)L;
  const int tid = threadIdx.x;
  __shared__ float red[8];
  float m = -3.0e38f;
  for (int i = tid; i < L; i += 256) m = fmaxf(m, p[i]);
  for (int o = 16; o > 0; o >>= 1) m = fmaxf(m, __shfl_xor(m, o, 32));
  if ((tid & 31) == 0) red[tid >> 5] = m;
  __syncthreads();
  float mx = red[0];
  for (int w = 1; w < 8; ++w) mx = fmaxf(mx, red[w]);
  __syncthreads();
  float s = 0.0f;
  for (int i = tid; i < L; i += 256) { float e = expf(p[i] - mx); p[i] = e; s += e; }
  for (int o = 16; o > 0; o >>= 1) s += __shfl_xor(s, o, 32);
  if ((tid & 31) == 0) red[tid >> 5] = s;
  __syncthreads();
  float tot = 0.0f;
  for (int w = 0; w < 8; ++w) tot += red[w];
  float inv = 1.0f / tot;
  for (int i = tid; i < L; i += 256) p[i] *= inv;
}

// y += D[h] * x   (per-channel scaled residual of the SSM branch)
__global__ void dx_add_kernel(float* __restrict__ y, const float* __restrict__ x,
                              const float* __restrict__ D, long long n)
{
  long long i = (long long)blockIdx.x * blockDim.x + threadIdx.x;
  if (i >= n) return;
  y[i] += D[i % HH] * x[i];
}

// cat = [x2, combined/3]; also writes combined/3 back in place.
__global__ void concat_kernel(const float* __restrict__ x2, float* __restrict__ combined,
                              float* __restrict__ cat, long long n)
{
  long long i = (long long)blockIdx.x * blockDim.x + threadIdx.x;
  if (i >= n) return;
  long long row = i / HH;
  int h = (int)(i % HH);
  float c = combined[i] * (1.0f / 3.0f);
  combined[i] = c;
  cat[row * (2 * HH) + h]      = x2[i];
  cat[row * (2 * HH) + HH + h] = c;
}

// x3 = x2 + gw*x2 + (1-gw)*combined
__global__ void gate_combine_kernel(const float* __restrict__ x2, const float* __restrict__ combined,
                                    const float* __restrict__ gw, float* __restrict__ x3, long long n)
{
  long long i = (long long)blockIdx.x * blockDim.x + threadIdx.x;
  if (i >= n) return;
  float g = gw[i];
  x3[i] = x2[i] + g * x2[i] + (1.0f - g) * combined[i];
}

// ---------------------------------------------------------------------------
static void gemm(hipStream_t st,
                 const float* A, int lda, long long sAo, long long sAi,
                 const float* B, int ldb, long long sBo, long long sBi,
                 float* C, int ldc, long long sCo, long long sCi,
                 const float* bias, int M, int N, int K,
                 int batches, int innerCount,
                 float alpha, int transB, int act, int accum)
{
  dim3 g(N / 64, M / 128, batches), b(256);
  if (transB == 0)
    gemm_wmma_kernel<0><<<g, b, 0, st>>>(A, lda, sAo, sAi, B, ldb, sBo, sBi,
                                         C, ldc, sCo, sCi, bias,
                                         M, N, K, innerCount, alpha, act, accum);
  else
    gemm_wmma_kernel<1><<<g, b, 0, st>>>(A, lda, sAo, sAi, B, ldb, sBo, sBi,
                                         C, ldc, sCo, sCi, bias,
                                         M, N, K, innerCount, alpha, act, accum);
}

extern "C" void kernel_launch(void* const* d_in, const int* in_sizes, int n_in,
                              void* d_out, int out_size, void* d_ws, size_t ws_size,
                              hipStream_t stream)
{
  (void)in_sizes; (void)n_in; (void)out_size; (void)ws_size;
  const float* x        = (const float*)d_in[0];
  const float* mem0     = (const float*)d_in[1];
  const float* mem1     = (const float*)d_in[2];
  const float* mem2     = (const float*)d_in[3];
  const float* A_log    = (const float*)d_in[4];
  const float* B_w      = (const float*)d_in[5];
  const float* C_w      = (const float*)d_in[6];
  const float* Dv       = (const float*)d_in[7];
  const float* sgate_w  = (const float*)d_in[8];
  const float* sgate_b  = (const float*)d_in[9];
  const float* outp_w   = (const float*)d_in[10];
  const float* outp_b   = (const float*)d_in[11];
  const float* sln_g    = (const float*)d_in[12];
  const float* sln_b    = (const float*)d_in[13];
  const float* attn_in_w  = (const float*)d_in[14];
  const float* attn_in_b  = (const float*)d_in[15];
  const float* attn_out_w = (const float*)d_in[16];
  const float* attn_out_b = (const float*)d_in[17];
  const float* cw1[3] = {(const float*)d_in[18], (const float*)d_in[22], (const float*)d_in[26]};
  const float* cb1[3] = {(const float*)d_in[19], (const float*)d_in[23], (const float*)d_in[27]};
  const float* cw2[3] = {(const float*)d_in[20], (const float*)d_in[24], (const float*)d_in[28]};
  const float* cb2[3] = {(const float*)d_in[21], (const float*)d_in[25], (const float*)d_in[29]};
  const float* retr_in_w  = (const float*)d_in[30];
  const float* retr_in_b  = (const float*)d_in[31];
  const float* retr_out_w = (const float*)d_in[32];
  const float* retr_out_b = (const float*)d_in[33];
  const float* mg_w   = (const float*)d_in[34];
  const float* mg_b   = (const float*)d_in[35];
  const float* ffn_w1 = (const float*)d_in[36];
  const float* ffn_b1 = (const float*)d_in[37];
  const float* ffn_w2 = (const float*)d_in[38];
  const float* ffn_b2 = (const float*)d_in[39];
  const float* n1_g = (const float*)d_in[40];
  const float* n1_b = (const float*)d_in[41];
  const float* n2_g = (const float*)d_in[42];
  const float* n2_b = (const float*)d_in[43];
  const float* n3_g = (const float*)d_in[44];
  const float* n3_b = (const float*)d_in[45];
  float* out = (float*)d_out;

  // ---- workspace arena (floats), with reuse ------------------------------
  float* wsf = (float*)d_ws;
  size_t offw = 0;
  auto alloc = [&](size_t n) { float* p = wsf + offw; offw += n; return p; };
  const size_t NBH = (size_t)BT * HH;                       // 8M elems
  float* gateb  = alloc((size_t)BT * SS);
  float* bxb    = alloc((size_t)BT * SS);
  float* states = alloc((size_t)BT * SS);
  float* buf1   = alloc(NBH);                               // y / attn-proj / gw / ffn-out
  float* buf2   = alloc(NBH);                               // ln-tmp / attnO / retrO
  float* xstate = alloc(NBH);                               // x_state / retrQ
  float* x1     = alloc(NBH);                               // x1 / x3
  float* x2     = alloc(NBH);
  float* comp_h = alloc((size_t)BSZ * 256 * 1024);
  float* comp_c = alloc((size_t)BSZ * 256 * HH);
  float* retrK  = alloc((size_t)BSZ * 256 * HH);
  float* retrV  = alloc((size_t)BSZ * 256 * HH);
  float* retrsc = alloc((size_t)BSZ * RHH * TT * 256);
  float* combined = alloc(NBH);
  float* big    = alloc((size_t)BT * 3 * HH + (size_t)NHH * TT * TT);
  float* qkv    = big;                                      // 24M
  float* scbuf  = big + (size_t)BT * 3 * HH;                // 16M (per-batch scores)
  float* cat2   = big;                                      // reused after attention
  float* ffnh   = big;                                      // reused after gating
  float* retrQ  = xstate;
  float* attnO  = buf2;
  float* retrO  = buf2;
  float* gw     = buf1;
  float* x3     = x1;
  const int EW = (int)((NBH + 255) / 256);

  // ---- 1) SSM branch -----------------------------------------------------
  gemm(stream, x, HH,0,0, sgate_w, HH,0,0, gateb, SS,0,0, sgate_b,
       BT, SS, HH, 1,1, 1.0f, 0, /*sigmoid*/3, 0);
  gemm(stream, x, HH,0,0, B_w, HH,0,0, bxb, SS,0,0, nullptr,
       BT, SS, HH, 1,1, 1.0f, 0, 0, 0);
  scan_kernel<<<dim3((BSZ * SS + 255) / 256), dim3(256), 0, stream>>>(gateb, bxb, A_log, states);
  gemm(stream, states, SS,0,0, C_w, SS,0,0, buf1, HH,0,0, nullptr,
       BT, HH, SS, 1,1, 1.0f, 0, 0, 0);
  dx_add_kernel<<<EW, 256, 0, stream>>>(buf1, x, Dv, (long long)NBH);
  ln_kernel<<<BT, 256, 0, stream>>>(buf1, x, sln_g, sln_b, buf2);
  gemm(stream, buf2, HH,0,0, outp_w, HH,0,0, xstate, HH,0,0, outp_b,
       BT, HH, HH, 1,1, 1.0f, 0, 0, 0);
  ln_kernel<<<BT, 256, 0, stream>>>(x, xstate, n1_g, n1_b, x1);

  // ---- 2) self-attention (16 heads, dh=64), per-batch scores buffer ------
  gemm(stream, x1, HH,0,0, attn_in_w, HH,0,0, qkv, 3*HH,0,0, attn_in_b,
       BT, 3*HH, HH, 1,1, 1.0f, 0, 0, 0);
  for (int b = 0; b < BSZ; ++b) {
    const float* qb = qkv + (size_t)b * TT * 3 * HH;
    gemm(stream, qb,       3*HH, 0, 64, qb + HH, 3*HH, 0, 64,
         scbuf, TT, 0, (long long)TT * TT, nullptr,
         TT, TT, 64, NHH, NHH, 0.125f, 0, 0, 0);
    softmax_kernel<<<NHH * TT, 256, 0, stream>>>(scbuf, TT);
    gemm(stream, scbuf, TT, 0, (long long)TT * TT, qb + 2*HH, 3*HH, 0, 64,
         attnO + (size_t)b * TT * HH, HH, 0, 64, nullptr,
         TT, 64, TT, NHH, NHH, 1.0f, /*transB*/1, 0, 0);
  }
  gemm(stream, attnO, HH,0,0, attn_out_w, HH,0,0, buf1, HH,0,0, attn_out_b,
       BT, HH, HH, 1,1, 1.0f, 0, 0, 0);
  ln_kernel<<<BT, 256, 0, stream>>>(x1, buf1, n2_g, n2_b, x2);

  // ---- 3) compressors + retrieval attention (4 heads, dh=256) ------------
  gemm(stream, x2, HH,0,0, retr_in_w, HH,0,0, retrQ, HH,0,0, retr_in_b,
       BT, HH, HH, 1,1, 1.0f, 0, 0, 0);
  const float* mems[3] = {mem0, mem1, mem2};
  const int memR[3] = {256, 512, 1024};
  const int cdim[3] = {1024, 512, 256};
  for (int i = 0; i < 3; ++i) {
    const float* mbase = mems[i] + (size_t)(memR[i] - 256) * HH;   // last 256 rows
    gemm(stream, mbase, HH, (long long)memR[i] * HH, 0, cw1[i], HH, 0, 0,
         comp_h, cdim[i], (long long)256 * cdim[i], 0, cb1[i],
         256, cdim[i], HH, BSZ, 1, 1.0f, 0, /*relu*/1, 0);
    gemm(stream, comp_h, cdim[i],0,0, cw2[i], cdim[i],0,0, comp_c, HH,0,0, cb2[i],
         BSZ * 256, HH, cdim[i], 1,1, 1.0f, 0, 0, 0);
    gemm(stream, comp_c, HH,0,0, retr_in_w + (size_t)HH * HH, HH,0,0, retrK, HH,0,0,
         retr_in_b + HH, BSZ * 256, HH, HH, 1,1, 1.0f, 0, 0, 0);
    gemm(stream, comp_c, HH,0,0, retr_in_w + (size_t)2 * HH * HH, HH,0,0, retrV, HH,0,0,
         retr_in_b + 2 * HH, BSZ * 256, HH, HH, 1,1, 1.0f, 0, 0, 0);
    gemm(stream, retrQ, HH, (long long)TT * HH, 256,
         retrK, HH, (long long)256 * HH, 256,
         retrsc, 256, (long long)RHH * TT * 256, (long long)TT * 256, nullptr,
         TT, 256, 256, BSZ * RHH, RHH, 0.0625f, 0, 0, 0);
    softmax_kernel<<<BSZ * RHH * TT, 256, 0, stream>>>(retrsc, 256);
    gemm(stream, retrsc, 256, (long long)RHH * TT * 256, (long long)TT * 256,
         retrV, HH, (long long)256 * HH, 256,
         retrO, HH, (long long)TT * HH, 256, nullptr,
         TT, 256, 256, BSZ * RHH, RHH, 1.0f, /*transB*/1, 0, 0);
    gemm(stream, retrO, HH,0,0, retr_out_w, HH,0,0, combined, HH,0,0, retr_out_b,
         BT, HH, HH, 1,1, 1.0f, 0, 0, /*accum*/(i > 0) ? 1 : 0);
  }

  // ---- 4) memory gating + FFN -------------------------------------------
  concat_kernel<<<EW, 256, 0, stream>>>(x2, combined, cat2, (long long)NBH);
  gemm(stream, cat2, 2*HH,0,0, mg_w, 2*HH,0,0, gw, HH,0,0, mg_b,
       BT, HH, 2*HH, 1,1, 1.0f, 0, /*sigmoid*/3, 0);
  gate_combine_kernel<<<EW, 256, 0, stream>>>(x2, combined, gw, x3, (long long)NBH);
  gemm(stream, x3, HH,0,0, ffn_w1, HH,0,0, ffnh, 4*HH,0,0, ffn_b1,
       BT, 4*HH, HH, 1,1, 1.0f, 0, /*gelu*/2, 0);
  gemm(stream, ffnh, 4*HH,0,0, ffn_w2, 4*HH,0,0, buf1, HH,0,0, ffn_b2,
       BT, HH, 4*HH, 1,1, 1.0f, 0, 0, 0);
  ln_kernel<<<BT, 256, 0, stream>>>(x3, buf1, n3_g, n3_b, out);
}
